// VectorQuantizer_59614146068928
// MI455X (gfx1250) — compile-verified, matrix operand order checked by reading, not run on hardware
//
#include <hip/hip_runtime.h>
#include <hip/hip_bf16.h>

// ---------------------------------------------------------------------------
// VQ-VAE vector quantizer for MI455X (gfx1250, wave32, WMMA + async LDS DMA).
// Outputs (concatenated in d_out, float): z_q [N*256], loss, commitment,
// codebook, idx [N] (as float).
// ---------------------------------------------------------------------------

typedef _Float16 v16h __attribute__((ext_vector_type(16)));
typedef _Float16 v8h  __attribute__((ext_vector_type(8)));
typedef float    v8f  __attribute__((ext_vector_type(8)));
typedef int      v4i  __attribute__((ext_vector_type(4)));

#define GLOBAL_AS __attribute__((address_space(1)))
#define LDS_AS    __attribute__((address_space(3)))

#define D_TOK 256             // token size
#define WAVES_PER_BLK 8
#define ROWS_PER_WAVE 16
#define ROWS_PER_BLK  (WAVES_PER_BLK * ROWS_PER_WAVE)   // 128
#define TILE_N 16             // codes per tile
#define LDS_ROW_H 264         // halves per staged code row (256 + 8 pad = 528B)
#define TILE_LDS_H (TILE_N * LDS_ROW_H)                 // 4224 halves / tile
#define NBUF 4                // LDS tile ring (prefetch depth 2)
#define NSPLIT 4              // codebook splits (grid.y) for occupancy

#if __has_builtin(__builtin_amdgcn_global_load_async_to_lds_b128) && \
    __has_builtin(__builtin_amdgcn_s_wait_asynccnt)
#define USE_ASYNC_LDS 1
#else
#define USE_ASYNC_LDS 0
#endif

union AFrag { v16h v; _Float16 h[16]; };
union BFrag { v16h v; v8h h2[2]; };

// ---- Phase 1: embedding f32 -> f16 (row-major) + exact ||e||^2 in f32 -----
__global__ void __launch_bounds__(256)
prep_codebook(const float* __restrict__ emb, _Float16* __restrict__ eh,
              float* __restrict__ e2) {
  const int c = blockIdx.x;          // code id
  const int d = threadIdx.x;         // 0..255
  const float v = emb[(size_t)c * D_TOK + d];
  eh[(size_t)c * D_TOK + d] = (_Float16)v;

  float s = v * v;
  #pragma unroll
  for (int m = 16; m >= 1; m >>= 1) s += __shfl_xor(s, m, 32);
  __shared__ float ss[WAVES_PER_BLK];
  if ((d & 31) == 0) ss[d >> 5] = s;
  __syncthreads();
  if (d == 0) {
    float t = 0.f;
    #pragma unroll
    for (int w = 0; w < WAVES_PER_BLK; ++w) t += ss[w];
    e2[c] = t;
  }
}

// Stage one 16x256 f16 code tile (8KB) into LDS: 256 threads x 2 x 16B.
__device__ __forceinline__ void
prefetch_tile(const _Float16* __restrict__ eh, _Float16* lbuf, int tile,
              int tid) {
  const int r = tid >> 4;               // code row within tile (0..15)
  const int j = (tid & 15) * 16;        // half offset within row (step 16)
  const _Float16* src = eh + (size_t)(tile * TILE_N + r) * D_TOK + j;
  _Float16* dst = lbuf + r * LDS_ROW_H + j;
#if USE_ASYNC_LDS
  __builtin_amdgcn_global_load_async_to_lds_b128(
      (GLOBAL_AS v4i*)src, (LDS_AS v4i*)dst, 0, 0);
  __builtin_amdgcn_global_load_async_to_lds_b128(
      (GLOBAL_AS v4i*)(src + 8), (LDS_AS v4i*)(dst + 8), 0, 0);
#else
  *(v8h*)dst       = *(const v8h*)src;
  *(v8h*)(dst + 8) = *(const v8h*)(src + 8);
#endif
}

// ---- Phase 2: WMMA GEMM (scores = ||e||^2 - 2 z.e) fused with argmin ------
// grid.x: row blocks (128 rows each); grid.y: codebook split (NSPLIT ways).
// Each block sweeps tilesPerSplit=K/16/NSPLIT code tiles and emits per-row
// partial (score, idx); merge_argmin reduces the splits.
__global__ void __launch_bounds__(256)
vq_argmin(const float* __restrict__ z, const _Float16* __restrict__ eh,
          const float* __restrict__ e2, float* __restrict__ bdPart,
          int* __restrict__ biPart, int tilesPerSplit, int nTok) {
  __shared__ _Float16 lds[NBUF * TILE_LDS_H];   // 4 x 8.25KB tile ring

  const int tid   = threadIdx.x;
  const int lane  = tid & 31;
  const int wave  = tid >> 5;
  const int mloc  = lane & 15;          // A row / B col / C col within tile
  const int khalf = (lane >> 4) << 4;   // 0 or 16: K-half owned by this lane
  const int rowBase = blockIdx.x * ROWS_PER_BLK + wave * ROWS_PER_WAVE;
  const int row     = rowBase + mloc;
  const int split   = blockIdx.y;
  const int tile0   = split * tilesPerSplit;

  // Load + convert this wave's 16 z rows into 8 resident A fragments.
  // A fragment (16-bit, 16x32): here each lane holds 16 contiguous K halves of its row (B uses the same order).
  AFrag a[8];
  const float* zr = z + (size_t)row * D_TOK;
  #pragma unroll
  for (int c = 0; c < 8; ++c) {
    float t[16];
    const float4* p = (const float4*)(zr + c * 32 + khalf);
    *(float4*)(t + 0)  = p[0];
    *(float4*)(t + 4)  = p[1];
    *(float4*)(t + 8)  = p[2];
    *(float4*)(t + 12) = p[3];
    #pragma unroll
    for (int i = 0; i < 16; ++i) a[c].h[i] = (_Float16)t[i];
  }

  float bd[8];
  int   bi[8];
  #pragma unroll
  for (int v = 0; v < 8; ++v) { bd[v] = 3.4e38f; bi[v] = 0; }

  // Prime the async pipeline two tiles deep.
  prefetch_tile(eh, lds, tile0, tid);
  if (tilesPerSplit > 1) prefetch_tile(eh, lds + TILE_LDS_H, tile0 + 1, tid);

  for (int t = 0; t < tilesPerSplit; ++t) {
    const int n = (tile0 + t) * TILE_N + mloc;   // code handled by this lane
    const float e2n = e2[n];

    // Issue DMA for tile t+2 (its buffer was last read at tile t-2; the
    // barrier below in iteration t-1 made that read globally complete).
    if (t + 2 < tilesPerSplit)
      prefetch_tile(eh, lds + ((t + 2) & (NBUF - 1)) * TILE_LDS_H,
                    tile0 + t + 2, tid);

#if USE_ASYNC_LDS
    if (t + 2 < tilesPerSplit)      __builtin_amdgcn_s_wait_asynccnt(4);
    else if (t + 1 < tilesPerSplit) __builtin_amdgcn_s_wait_asynccnt(2);
    else                            __builtin_amdgcn_s_wait_asynccnt(0);
#endif
    __syncthreads();   // tile t fully staged; previous buffers retired

    // B fragments from LDS: lane = column n; 2 x 16B reads per K-chunk.
    const _Float16* lrow =
        lds + (t & (NBUF - 1)) * TILE_LDS_H + mloc * LDS_ROW_H + khalf;
    BFrag b[8];
    #pragma unroll
    for (int c = 0; c < 8; ++c) {
      b[c].h2[0] = *(const v8h*)(lrow + c * 32);
      b[c].h2[1] = *(const v8h*)(lrow + c * 32 + 8);
    }

    v8f acc = {0.f, 0.f, 0.f, 0.f, 0.f, 0.f, 0.f, 0.f};
    #pragma unroll
    for (int c = 0; c < 8; ++c)
      acc = __builtin_amdgcn_wmma_f32_16x16x32_f16(
          false, a[c].v, false, b[c].v, (short)0, acc, false, false);

    // C layout: VGPR v -> M = v (lanes 0-15) / v+8 (lanes 16-31), N = lane&15
    #pragma unroll
    for (int v = 0; v < 8; ++v) {
      const float s = fmaf(-2.0f, acc[v], e2n);  // ||e||^2 - 2 z.e
      if (s < bd[v]) { bd[v] = s; bi[v] = n; }
    }
  }

  // Min-reduce each row's candidates across its 16-lane half (xor 1,2,4,8
  // never crosses the 16-lane boundary in wave32).
  #pragma unroll
  for (int v = 0; v < 8; ++v) {
    float d = bd[v]; int i = bi[v];
    #pragma unroll
    for (int m = 1; m < 16; m <<= 1) {
      const float od = __shfl_xor(d, m, 32);
      const int   oi = __shfl_xor(i, m, 32);
      if (od < d || (od == d && oi < i)) { d = od; i = oi; }
    }
    bd[v] = d; bi[v] = i;
  }
  if (mloc == 0) {                  // lane 0 -> rows +0..7, lane 16 -> +8..15
    const int mbase = rowBase + (lane >> 4) * 8;
    #pragma unroll
    for (int v = 0; v < 8; ++v) {
      bdPart[(size_t)split * nTok + mbase + v] = bd[v];
      biPart[(size_t)split * nTok + mbase + v] = bi[v];
    }
  }
}

// ---- Phase 2b: merge per-split partials (argmin, ties -> lower index) -----
__global__ void __launch_bounds__(256)
merge_argmin(const float* __restrict__ bdPart, const int* __restrict__ biPart,
             int* __restrict__ bestIdx, int nTok) {
  const int r = blockIdx.x * 256 + threadIdx.x;
  if (r >= nTok) return;
  float d = bdPart[r];
  int   i = biPart[r];
  #pragma unroll
  for (int s = 1; s < NSPLIT; ++s) {
    const float od = bdPart[(size_t)s * nTok + r];
    const int   oi = biPart[(size_t)s * nTok + r];
    if (od < d || (od == d && oi < i)) { d = od; i = oi; }
  }
  bestIdx[r] = i;
}

// ---- Phase 3: gather z_q (exact f32), idx->float, per-token loss partial ---
__global__ void __launch_bounds__(256)
gather_loss(const float* __restrict__ z, const float* __restrict__ emb,
            const int* __restrict__ bestIdx, float* __restrict__ zq,
            float* __restrict__ idxf, double* __restrict__ partials) {
  const int t = blockIdx.x;
  const int d = threadIdx.x;
  const int i = bestIdx[t];
  const float e  = emb[(size_t)i * D_TOK + d];
  const float zz = z[(size_t)t * D_TOK + d];
  zq[(size_t)t * D_TOK + d] = e;                 // z_q_st forward value == z_q
  const float diff = e - zz;

  __shared__ double sm[256];
  sm[d] = (double)diff * (double)diff;
  __syncthreads();
  #pragma unroll
  for (int s = 128; s > 0; s >>= 1) {
    if (d < s) sm[d] += sm[d + s];
    __syncthreads();
  }
  if (d == 0) { partials[t] = sm[0]; idxf[t] = (float)i; }
}

// ---- Phase 4: deterministic fixed-order reduction of partial sums ----------
__global__ void __launch_bounds__(256)
finalize(const double* __restrict__ partials, float* __restrict__ lossOut,
         int nTok) {
  __shared__ double sm[256];
  double s = 0.0;
  for (int t = threadIdx.x; t < nTok; t += 256) s += partials[t];
  sm[threadIdx.x] = s;
  __syncthreads();
  #pragma unroll
  for (int k = 128; k > 0; k >>= 1) {
    if (threadIdx.x < k) sm[threadIdx.x] += sm[threadIdx.x + k];
    __syncthreads();
  }
  if (threadIdx.x == 0) {
    const double mse = sm[0] / ((double)nTok * (double)D_TOK);
    lossOut[0] = (float)(1.25 * mse);   // loss = commitment + codebook
    lossOut[1] = (float)(0.25 * mse);   // commitment
    lossOut[2] = (float)mse;            // codebook
  }
}

extern "C" void kernel_launch(void* const* d_in, const int* in_sizes, int n_in,
                              void* d_out, int out_size, void* d_ws,
                              size_t ws_size, hipStream_t stream) {
  (void)n_in; (void)out_size; (void)ws_size;
  const float* z   = (const float*)d_in[0];   // [32,1024,256] f32
  const float* emb = (const float*)d_in[1];   // [8192,256] f32

  const int N = in_sizes[0] / D_TOK;          // 32768 tokens
  const int K = in_sizes[1] / D_TOK;          // 8192 codes

  // Workspace layout (all offsets >=32B aligned):
  //   [0)       partials : double[N]          (262144 B)
  //   then      e2       : float[K]           (32768 B)
  //   then      bestIdx  : int[N]             (131072 B)
  //   then      bdPart   : float[NSPLIT*N]    (524288 B)
  //   then      biPart   : int[NSPLIT*N]      (524288 B)
  //   then      eh       : f16[K*256]         (4 MB)
  char* ws = (char*)d_ws;
  double*   partials = (double*)ws;
  float*    e2       = (float*)(ws + (size_t)N * sizeof(double));
  int*      bestIdx  = (int*)((char*)e2 + (size_t)K * sizeof(float));
  float*    bdPart   = (float*)((char*)bestIdx + (size_t)N * sizeof(int));
  int*      biPart   = (int*)((char*)bdPart + (size_t)NSPLIT * N * sizeof(float));
  _Float16* eh       = (_Float16*)((char*)biPart + (size_t)NSPLIT * N * sizeof(int));

  float* out    = (float*)d_out;
  float* zq     = out;                        // [N*256]
  float* losses = out + (size_t)N * D_TOK;    // loss, commitment, codebook
  float* idxf   = losses + 3;                 // [N] indices as float

  prep_codebook<<<K, 256, 0, stream>>>(emb, eh, e2);
  dim3 grid(N / ROWS_PER_BLK, NSPLIT);
  vq_argmin<<<grid, 256, 0, stream>>>(z, eh, e2, bdPart, biPart,
                                      K / TILE_N / NSPLIT, N);
  merge_argmin<<<(N + 255) / 256, 256, 0, stream>>>(bdPart, biPart, bestIdx, N);
  gather_loss<<<N, 256, 0, stream>>>(z, emb, bestIdx, zq, idxf, partials);
  finalize<<<1, 256, 0, stream>>>(partials, losses, N);
}
